// mtl_net_25658134626935
// MI455X (gfx1250) — matrix operand order repaired and checked by reading, not compiled, not run on hardware
//
#include <hip/hip_runtime.h>
#include <hip/hip_bf16.h>
#include <math.h>

// ---------------------------------------------------------------------------
// Problem constants (from the reference)
// ---------------------------------------------------------------------------
#define B_    64
#define T_    366
#define IN_   24
#define H1_   1024
#define H2_   2048
#define HM_   2048
#define PEN_  1024
#define C_    21
#define BT_   (B_ * T_)      // 23424, divisible by 64
#define KPAD_ 32             // pad IN=24 -> 32 for the K=32 bf16 WMMA

typedef __bf16 bf16_t;
typedef bf16_t v16bf __attribute__((ext_vector_type(16)));
typedef float  v8f   __attribute__((ext_vector_type(8)));

// ---------------------------------------------------------------------------
// WMMA fragment loaders (wave32, ISA §7.12.2 16-bit layouts)
//
// A (16x32 bf16, M x K):  lane L holds row M = L&15.
//   K = (lane>>4)*16 + e, e = 0..15 (matches B)
//
//   -> two contiguous 16-byte chunks at k0 + (lane>>4)*16 and +8.
//
// B (32x16 bf16, K x N), sourced from weights stored (N,K) row-major:
//   lane L holds column N = L&15; K = (lane>>4)*16 + e, e = 0..15
//   -> one contiguous 32-byte chunk from row n at k0 + (lane>>4)*16.
// ---------------------------------------------------------------------------
__device__ __forceinline__ v16bf load_a_frag(const bf16_t* __restrict__ A,
                                             int lda, int m0, int k0, int lane) {
    const int m  = m0 + (lane & 15);
    const int kh = lane >> 4;
    const bf16_t* p = A + (size_t)m * lda + k0 + kh * 16;
    union { v16bf v; uint4 q[2]; } f;
    f.q[0] = *(const uint4*)(p);
    f.q[1] = *(const uint4*)(p + 8);
    return f.v;
}

__device__ __forceinline__ v16bf load_b_frag(const bf16_t* __restrict__ W,
                                             int ldb, int n0, int k0, int lane) {
    const int n  = n0 + (lane & 15);
    const int kh = lane >> 4;
    const bf16_t* p = W + (size_t)n * ldb + k0 + kh * 16;
    return *(const v16bf*)p;   // 32 contiguous bytes, 32B-aligned
}

__device__ __forceinline__ v8f wmma_bf16(v16bf a, v16bf b, v8f c) {
    return __builtin_amdgcn_wmma_f32_16x16x32_bf16(
        /*neg_a=*/false, a, /*neg_b=*/false, b,
        /*c_mod=*/(short)0, c, /*reuse_a=*/false, /*reuse_b=*/false);
}

// ---------------------------------------------------------------------------
// Generic bf16 GEMM with f32 accumulate:
//   out[m,n] = act( sum_k A[m,k] * W[n,k] + bias[n] )
// A: (M,K) bf16 row-major.  W: (N,K) bf16 row-major (i.e. B = W^T).
// Block tile 64(M) x 256(N), 8 waves; each wave owns a 32x64 register tile
// (2 M-frags x 4 N-frags -> 8 f32 accumulators, 8 WMMAs per K-step).
// Requires M%64==0, N%256==0, K%32==0 (true for every call here).
// ---------------------------------------------------------------------------
__global__ __launch_bounds__(256)
void gemm_bf16_kernel(const bf16_t* __restrict__ A,
                      const bf16_t* __restrict__ W,
                      const float*  __restrict__ bias,
                      bf16_t* __restrict__ outB,   // bf16 output (or null)
                      float*  __restrict__ outF,   // f32  output (or null)
                      int M, int N, int K, int relu) {
    const int lane = threadIdx.x & 31;
    const int wave = threadIdx.x >> 5;
    const int wm = wave >> 2;     // 0..1  -> 32-row slab
    const int wn = wave & 3;      // 0..3  -> 64-col slab
    const int m0 = blockIdx.y * 64  + wm * 32;
    const int n0 = blockIdx.x * 256 + wn * 64;

    v8f acc[2][4] = {};

    for (int k0 = 0; k0 < K; k0 += 32) {
        if (k0 + 64 < K) {  // pull streaming weight rows toward the WGP
            __builtin_prefetch((const void*)(W + (size_t)(n0 + (lane & 15)) * K + k0 + 64), 0, 3);
        }
        v16bf a0 = load_a_frag(A, K, m0,      k0, lane);
        v16bf a1 = load_a_frag(A, K, m0 + 16, k0, lane);
        #pragma unroll
        for (int ni = 0; ni < 4; ++ni) {
            v16bf bfr = load_b_frag(W, K, n0 + ni * 16, k0, lane);
            acc[0][ni] = wmma_bf16(a0, bfr, acc[0][ni]);
            acc[1][ni] = wmma_bf16(a1, bfr, acc[1][ni]);
        }
    }

    // C/D layout: element r of v8f -> M = m0 + (lane>>4)*8 + r, N = n0 + (lane&15)
    #pragma unroll
    for (int mi = 0; mi < 2; ++mi) {
        #pragma unroll
        for (int ni = 0; ni < 4; ++ni) {
            const int mrow = m0 + mi * 16 + (lane >> 4) * 8;
            const int ncol = n0 + ni * 16 + (lane & 15);
            const float bv = bias ? bias[ncol] : 0.0f;
            #pragma unroll
            for (int r = 0; r < 8; ++r) {
                float v = acc[mi][ni][r] + bv;
                if (relu) v = fmaxf(v, 0.0f);
                const size_t idx = (size_t)(mrow + r) * N + ncol;
                if (outF) outF[idx] = v;
                else      outB[idx] = (bf16_t)v;
            }
        }
    }
}

// ---------------------------------------------------------------------------
// Fused GRU step (one launch per timestep t):
//   hg = h_prev @ W_hh^T (+ b_hh), gates, h_new; writes h_next and hs[:,t,:].
// Grid: HM/64 = 32 blocks; each block computes a 64(batch) x 64(col) tile of
// h_new.  Each wave computes 2 M-tiles x 1 N-tile for ALL THREE gates, so the
// gate math is purely wave-local (no cross-wave exchange needed).
// ---------------------------------------------------------------------------
__global__ __launch_bounds__(256)
void gru_step_kernel(const bf16_t* __restrict__ hprev,   // (B, HM) bf16
                     bf16_t* __restrict__ hnext,         // (B, HM) bf16
                     const bf16_t* __restrict__ Whh,     // (3*HM, HM) bf16
                     const float*  __restrict__ xg,      // (B*T, 3*HM) f32 (b_ih included)
                     const float*  __restrict__ b_hh,    // (3*HM)
                     bf16_t* __restrict__ hs,            // (B*T, HM) bf16
                     int t) {
    const int lane = threadIdx.x & 31;
    const int wave = threadIdx.x >> 5;
    const int nt    = wave & 3;          // N-tile within the 64-col block
    const int mbase = (wave >> 2) * 32;  // 0 or 32 (batch rows)
    const int jb    = blockIdx.x * 64;   // hidden-column base

    v8f acc[2][3] = {};

    for (int k0 = 0; k0 < HM_; k0 += 32) {
        if (k0 + 64 < HM_) {
            __builtin_prefetch((const void*)(Whh + (size_t)(jb + nt * 16 + (lane & 15)) * HM_ + k0 + 64), 0, 3);
        }
        v16bf a0 = load_a_frag(hprev, HM_, mbase,      k0, lane);
        v16bf a1 = load_a_frag(hprev, HM_, mbase + 16, k0, lane);
        #pragma unroll
        for (int g = 0; g < 3; ++g) {
            v16bf bfr = load_b_frag(Whh, HM_, g * HM_ + jb + nt * 16, k0, lane);
            acc[0][g] = wmma_bf16(a0, bfr, acc[0][g]);
            acc[1][g] = wmma_bf16(a1, bfr, acc[1][g]);
        }
    }

    const int j = jb + nt * 16 + (lane & 15);
    const float bhr = b_hh[j];
    const float bhz = b_hh[HM_ + j];
    const float bhn = b_hh[2 * HM_ + j];

    #pragma unroll
    for (int mi = 0; mi < 2; ++mi) {
        const int brow = mbase + mi * 16 + (lane >> 4) * 8;
        #pragma unroll
        for (int r = 0; r < 8; ++r) {
            const int b = brow + r;
            const size_t xrow = (size_t)(b * T_ + t) * (3 * HM_);
            const float hr = acc[mi][0][r] + bhr;
            const float hz = acc[mi][1][r] + bhz;
            const float hn = acc[mi][2][r] + bhn;
            const float xr = xg[xrow + j];
            const float xz = xg[xrow + HM_ + j];
            const float xn = xg[xrow + 2 * HM_ + j];
            const float rg = 1.0f / (1.0f + __expf(-(xr + hr)));
            const float zg = 1.0f / (1.0f + __expf(-(xz + hz)));
            const float ng = tanhf(xn + rg * hn);
            const float hp = (float)hprev[(size_t)b * HM_ + j];
            const float hv = (1.0f - zg) * ng + zg * hp;
            hnext[(size_t)b * HM_ + j]        = (bf16_t)hv;
            hs[(size_t)(b * T_ + t) * HM_ + j] = (bf16_t)hv;
        }
    }
}

// ---------------------------------------------------------------------------
// Per-sample indexed heads: y_h[b,t] = out_s[b,t,:] . W_h[label[b],:] + b_h[label[b]]
// One wave per (b,t) row, shfl reduction (wave32).
// ---------------------------------------------------------------------------
__global__ __launch_bounds__(256)
void heads_kernel(const bf16_t* __restrict__ outs,   // (B*T, PEN) bf16
                  const int* __restrict__ labels,
                  const float* __restrict__ W4, const float* __restrict__ b4,
                  const float* __restrict__ W5, const float* __restrict__ b5,
                  const float* __restrict__ W6, const float* __restrict__ b6,
                  float* __restrict__ y) {
    const int row  = blockIdx.x * 8 + (threadIdx.x >> 5);
    const int lane = threadIdx.x & 31;
    if (row >= BT_) return;
    const int b   = row / T_;
    const int lab = labels[b];
    const bf16_t* o  = outs + (size_t)row * PEN_;
    const float*  w4 = W4 + (size_t)lab * PEN_;
    const float*  w5 = W5 + (size_t)lab * PEN_;
    const float*  w6 = W6 + (size_t)lab * PEN_;
    float s4 = 0.f, s5 = 0.f, s6 = 0.f;
    for (int k = lane; k < PEN_; k += 32) {
        const float v = (float)o[k];
        s4 += v * w4[k];
        s5 += v * w5[k];
        s6 += v * w6[k];
    }
    #pragma unroll
    for (int off = 16; off > 0; off >>= 1) {
        s4 += __shfl_down(s4, off, 32);
        s5 += __shfl_down(s5, off, 32);
        s6 += __shfl_down(s6, off, 32);
    }
    if (lane == 0) {
        y[row]           = s4 + b4[lab];
        y[BT_ + row]     = s5 + b5[lab];
        y[2 * BT_ + row] = s6 + b6[lab];
    }
}

// ---------------------------------------------------------------------------
// Utility kernels
// ---------------------------------------------------------------------------
__global__ void cvt_f32_to_bf16(const float* __restrict__ s, bf16_t* __restrict__ d, size_t n) {
    size_t i = (size_t)blockIdx.x * blockDim.x + threadIdx.x;
    if (i < n) d[i] = (bf16_t)s[i];
}

// (rows,kin) f32 -> (rows,kout) bf16 with zero K-padding
__global__ void pad_convert(const float* __restrict__ s, bf16_t* __restrict__ d,
                            int rows, int kin, int kout) {
    size_t i = (size_t)blockIdx.x * blockDim.x + threadIdx.x;
    size_t total = (size_t)rows * kout;
    if (i >= total) return;
    int r = (int)(i / kout), k = (int)(i % kout);
    d[i] = (k < kin) ? (bf16_t)s[(size_t)r * kin + k] : (bf16_t)0.0f;
}

__global__ void fill_bf16(bf16_t* __restrict__ d, size_t n) {
    size_t i = (size_t)blockIdx.x * blockDim.x + threadIdx.x;
    if (i < n) d[i] = (bf16_t)0.0f;
}

__global__ void fill_f32(float* __restrict__ d, size_t n) {
    size_t i = (size_t)blockIdx.x * blockDim.x + threadIdx.x;
    if (i < n) d[i] = 0.0f;
}

// ---------------------------------------------------------------------------
// Launch
// ---------------------------------------------------------------------------
static inline char* bump(char*& p, size_t bytes) {
    char* r = p;
    p += (bytes + 255) & ~(size_t)255;
    return r;
}

extern "C" void kernel_launch(void* const* d_in, const int* in_sizes, int n_in,
                              void* d_out, int out_size, void* d_ws, size_t ws_size,
                              hipStream_t stream) {
    const float* x      = (const float*)d_in[0];
    const int*   labels = (const int*)  d_in[1];
    const float* W1     = (const float*)d_in[2];
    const float* b1     = (const float*)d_in[3];
    const float* W2     = (const float*)d_in[4];
    const float* b2     = (const float*)d_in[5];
    const float* W_ih   = (const float*)d_in[6];
    const float* W_hh   = (const float*)d_in[7];
    const float* b_ih   = (const float*)d_in[8];
    const float* b_hh   = (const float*)d_in[9];
    const float* W3     = (const float*)d_in[10];
    const float* b3     = (const float*)d_in[11];
    const float* W4     = (const float*)d_in[12];
    const float* b4     = (const float*)d_in[13];
    const float* W5     = (const float*)d_in[14];
    const float* b5     = (const float*)d_in[15];
    const float* W6     = (const float*)d_in[16];
    const float* b6     = (const float*)d_in[17];
    float* y = (float*)d_out;

    // ---- workspace layout (all 256B-aligned) ----
    char* p = (char*)d_ws;
    bf16_t* xpad  = (bf16_t*)bump(p, (size_t)BT_ * KPAD_ * 2);
    bf16_t* W1p   = (bf16_t*)bump(p, (size_t)H1_ * KPAD_ * 2);
    bf16_t* W2b   = (bf16_t*)bump(p, (size_t)H2_ * H1_ * 2);
    bf16_t* Wihb  = (bf16_t*)bump(p, (size_t)3 * HM_ * H2_ * 2);
    bf16_t* Whhb  = (bf16_t*)bump(p, (size_t)3 * HM_ * HM_ * 2);
    bf16_t* W3b   = (bf16_t*)bump(p, (size_t)PEN_ * HM_ * 2);
    bf16_t* h1    = (bf16_t*)bump(p, (size_t)BT_ * H1_ * 2);
    bf16_t* h2    = (bf16_t*)bump(p, (size_t)BT_ * H2_ * 2);
    float*  xg    = (float*) bump(p, (size_t)BT_ * 3 * HM_ * 4);
    bf16_t* hs    = (bf16_t*)bump(p, (size_t)BT_ * HM_ * 2);
    bf16_t* outs  = (bf16_t*)bump(p, (size_t)BT_ * PEN_ * 2);
    bf16_t* hbuf0 = (bf16_t*)bump(p, (size_t)B_ * HM_ * 2);
    bf16_t* hbuf1 = (bf16_t*)bump(p, (size_t)B_ * HM_ * 2);

    const int TB = 256;
    auto blocks = [](size_t n, int tb) { return (unsigned)((n + tb - 1) / tb); };

    // ---- weight / input conversion to bf16 (weights then live in L2: ~58 MB) ----
    pad_convert<<<blocks((size_t)BT_ * KPAD_, TB), TB, 0, stream>>>(x, xpad, BT_, IN_, KPAD_);
    pad_convert<<<blocks((size_t)H1_ * KPAD_, TB), TB, 0, stream>>>(W1, W1p, H1_, IN_, KPAD_);
    cvt_f32_to_bf16<<<blocks((size_t)H2_ * H1_, TB), TB, 0, stream>>>(W2, W2b, (size_t)H2_ * H1_);
    cvt_f32_to_bf16<<<blocks((size_t)3 * HM_ * H2_, TB), TB, 0, stream>>>(W_ih, Wihb, (size_t)3 * HM_ * H2_);
    cvt_f32_to_bf16<<<blocks((size_t)3 * HM_ * HM_, TB), TB, 0, stream>>>(W_hh, Whhb, (size_t)3 * HM_ * HM_);
    cvt_f32_to_bf16<<<blocks((size_t)PEN_ * HM_, TB), TB, 0, stream>>>(W3, W3b, (size_t)PEN_ * HM_);
    fill_bf16<<<blocks((size_t)B_ * HM_, TB), TB, 0, stream>>>(hbuf0, (size_t)B_ * HM_);
    fill_bf16<<<blocks((size_t)B_ * HM_, TB), TB, 0, stream>>>(hbuf1, (size_t)B_ * HM_);
    fill_f32<<<blocks((size_t)out_size, TB), TB, 0, stream>>>(y, (size_t)out_size);

    auto gemm = [&](const bf16_t* A, const bf16_t* W, const float* bias,
                    bf16_t* oB, float* oF, int M, int N, int K, int relu) {
        dim3 g(N / 256, M / 64);
        gemm_bf16_kernel<<<g, 256, 0, stream>>>(A, W, bias, oB, oF, M, N, K, relu);
    };

    // ---- feed-forward trunk ----
    gemm(xpad, W1p,  b1,   h1,  nullptr, BT_, H1_,    KPAD_, 1);   // L1 + relu
    gemm(h1,   W2b,  b2,   h2,  nullptr, BT_, H2_,    H1_,   1);   // L2 + relu
    gemm(h2,   Wihb, b_ih, nullptr, xg,  BT_, 3 * HM_, H2_,  0);   // GRU input gates (f32)

    // ---- sequential GRU recurrence (W_hh stays resident in L2) ----
    bf16_t* hA = hbuf0;
    bf16_t* hB = hbuf1;
    for (int t = 0; t < T_; ++t) {
        gru_step_kernel<<<dim3(HM_ / 64), 256, 0, stream>>>(hA, hB, Whhb, xg, b_hh, hs, t);
        bf16_t* tmp = hA; hA = hB; hB = tmp;
    }

    // ---- output projection + heads ----
    gemm(hs, W3b, b3, outs, nullptr, BT_, PEN_, HM_, 1);           // L3 + relu
    heads_kernel<<<blocks((size_t)BT_, 8), 256, 0, stream>>>(
        outs, labels, W4, b4, W5, b5, W6, b6, y);
}
